// FieldPaiNN_2972117369378
// MI455X (gfx1250) — matrix operand order repaired and checked by reading, not compiled, not run on hardware
//
#include <hip/hip_runtime.h>
#include <hip/hip_bf16.h>

// ---------------- problem constants (match reference) ----------------
#define F_DIM   128
#define TSTEPS  3
#define N_RBF   20
#define N_ATOMS 10000
#define N_EDGES 320000
#define N_MOL   16

// ---------------- CDNA5 WMMA types & helpers ----------------
typedef __attribute__((ext_vector_type(16))) __bf16         bf16x16;
typedef __attribute__((ext_vector_type(8)))  __bf16         bf16x8;
typedef __attribute__((ext_vector_type(8)))  float          f32x8;
typedef __attribute__((ext_vector_type(16))) unsigned short u16x16;

__device__ __forceinline__ unsigned short f2bf(float f) {
  unsigned int u = __float_as_uint(f);
  u += 0x7fffu + ((u >> 16) & 1u);       // round-to-nearest-even
  return (unsigned short)(u >> 16);
}

__device__ __forceinline__ f32x8 wmma_bf16(bf16x16 a, bf16x16 b, f32x8 c) {
  // v_wmma_f32_16x16x32_bf16 : D = A(16x32) * B(32x16) + C(16x16)
  return __builtin_amdgcn_wmma_f32_16x16x32_bf16(false, a, false, b, (short)0, c, false, false);
}

// B fragment: pre-packed fragment-major bf16; 32 contiguous bytes per lane.
__device__ __forceinline__ bf16x16 loadB(const unsigned short* __restrict__ Wp, int tile, int lane) {
  const bf16x8* p = (const bf16x8*)(Wp + (size_t)tile * 512 + lane * 16);
  bf16x8 lo = p[0], hi = p[1];
  return __builtin_shufflevector(lo, hi, 0,1,2,3,4,5,6,7,8,9,10,11,12,13,14,15);
}

// A fragment from row-major fp32 (rowptr = &in[row*ld + kt*32]); per ISA
// 16-bit A layout: lanes<16 hold K {0..7,16..23}, lanes>=16 hold K {8..15,24..31}.
__device__ __forceinline__ bf16x16 makeA_f32(const float* __restrict__ rowptr, int half) {
  const float* p0 = rowptr + half * 8;
  const float* p1 = rowptr + half * 8 + 16;
  u16x16 u;
#pragma unroll
  for (int i = 0; i < 8; ++i) u[i] = f2bf(p0[i]);
#pragma unroll
  for (int i = 0; i < 8; ++i) u[8 + i] = f2bf(p1[i]);
  return __builtin_bit_cast(bf16x16, u);
}

// A fragment from LDS bf16 row (rowptr = &hid[row*128 + kt*32]).
__device__ __forceinline__ bf16x16 makeA_lds(const unsigned short* __restrict__ rowptr, int half) {
  const bf16x8* p0 = (const bf16x8*)(rowptr + half * 8);
  const bf16x8* p1 = (const bf16x8*)(rowptr + half * 8 + 16);
  return __builtin_shufflevector(p0[0], p1[0], 0,1,2,3,4,5,6,7,8,9,10,11,12,13,14,15);
}

// ---------------- weight pre-pack (fp32 -> fragment-major bf16) ----------------
__global__ void pack_kernel(const float* __restrict__ src, int src_ld, int col0, int Kreal,
                            int Ktiles, int Ntiles, unsigned short* __restrict__ dst) {
  int tid = blockIdx.x * blockDim.x + threadIdx.x;
  int total = Ktiles * Ntiles * 512;
  if (tid >= total) return;
  int kt = tid / (Ntiles * 512);
  int rr = tid - kt * Ntiles * 512;
  int nt = rr / 512;
  int s  = rr - nt * 512;
  int lane = s >> 4, i = s & 15;
  int K   = kt * 32 + ((i >> 3) << 4) + ((lane >> 4) << 3) + (i & 7);   // B layout: element i <-> K = 8*half+i (i<8), 16+8*half+i-8 (i>=8)
  int col = nt * 16 + (lane & 15);
  float v = (K < Kreal) ? src[(size_t)K * src_ld + col0 + col] : 0.0f;
  dst[tid] = f2bf(v);
}

// ---------------- init: q = emb[z], mu = 0, Eat = e_field[idx_m] ----------------
__global__ void init_kernel(const int* __restrict__ z, const int* __restrict__ idx_m,
                            const float* __restrict__ emb, const float* __restrict__ e_field,
                            float* __restrict__ q, float* __restrict__ mu, float* __restrict__ Eat) {
  int tid = blockIdx.x * blockDim.x + threadIdx.x;   // N*128
  int n = tid >> 7, f = tid & 127;
  q[tid] = emb[(size_t)z[n] * 128 + f];
  mu[((size_t)n * 3 + 0) * 128 + f] = 0.0f;
  mu[((size_t)n * 3 + 1) * 128 + f] = 0.0f;
  mu[((size_t)n * 3 + 2) * 128 + f] = 0.0f;
  if (f < 3) Eat[n * 3 + f] = e_field[idx_m[n] * 3 + f];
}

// ---------------- 2-layer MLP: out = silu(in@W1+b1)@W2+b2, one wave / 16 rows ----------------
template <int KIN>
__global__ __launch_bounds__(32) void mlp16_kernel(const float* __restrict__ in,
                                                   const unsigned short* __restrict__ W1p,
                                                   const float* __restrict__ b1,
                                                   const unsigned short* __restrict__ W2p,
                                                   const float* __restrict__ b2, int Nout,
                                                   float* __restrict__ out) {
  constexpr int KT1 = KIN / 32;
  __shared__ unsigned short hid[16 * 128];           // bf16 hidden activations
  const int lane = threadIdx.x;
  const int r = lane & 15, half = lane >> 4;
  const int row0 = blockIdx.x * 16;
  const float* rowbase = in + (size_t)(row0 + r) * KIN;
  bf16x16 af[KT1];
#pragma unroll
  for (int kt = 0; kt < KT1; ++kt) af[kt] = makeA_f32(rowbase + kt * 32, half);
#pragma unroll
  for (int ct = 0; ct < 8; ++ct) {                   // hidden = 128 -> 8 column tiles
    float bias = b1[ct * 16 + r];
    f32x8 acc = {bias, bias, bias, bias, bias, bias, bias, bias};
#pragma unroll
    for (int kt = 0; kt < KT1; ++kt) acc = wmma_bf16(af[kt], loadB(W1p, kt * 8 + ct, lane), acc);
#pragma unroll
    for (int v = 0; v < 8; ++v) {                    // C layout: row = v + 8*half, col = r
      float h = acc[v];
      h = h * (1.0f / (1.0f + __expf(-h)));          // SiLU
      hid[(v + half * 8) * 128 + ct * 16 + r] = f2bf(h);
    }
  }
  __syncthreads();
  const int NT2 = Nout >> 4;
  for (int ct2 = 0; ct2 < NT2; ++ct2) {
    float bias = b2[ct2 * 16 + r];
    f32x8 acc = {bias, bias, bias, bias, bias, bias, bias, bias};
#pragma unroll
    for (int kt = 0; kt < 4; ++kt) {
      bf16x16 a2 = makeA_lds(&hid[r * 128 + kt * 32], half);
      acc = wmma_bf16(a2, loadB(W2p, kt * NT2 + ct2, lane), acc);
    }
#pragma unroll
    for (int v = 0; v < 8; ++v)
      out[(size_t)(row0 + v + half * 8) * Nout + ct2 * 16 + r] = acc[v];
  }
}

// ---------------- single GEMM: out = in@W (K=128, no bias), one wave / 16 rows ----------------
__global__ __launch_bounds__(32) void linear16_kernel(const float* __restrict__ in,
                                                      const unsigned short* __restrict__ Wp,
                                                      int Nout, float* __restrict__ out) {
  const int lane = threadIdx.x;
  const int r = lane & 15, half = lane >> 4;
  const int row0 = blockIdx.x * 16;
  const float* rowbase = in + (size_t)(row0 + r) * 128;
  bf16x16 af[4];
#pragma unroll
  for (int kt = 0; kt < 4; ++kt) af[kt] = makeA_f32(rowbase + kt * 32, half);
  const int NT = Nout >> 4;
  for (int ct = 0; ct < NT; ++ct) {
    f32x8 acc = {0.f, 0.f, 0.f, 0.f, 0.f, 0.f, 0.f, 0.f};
#pragma unroll
    for (int kt = 0; kt < 4; ++kt) acc = wmma_bf16(af[kt], loadB(Wp, kt * NT + ct, lane), acc);
#pragma unroll
    for (int v = 0; v < 8; ++v)
      out[(size_t)(row0 + v + half * 8) * Nout + ct * 16 + r] = acc[v];
  }
}

// ---------------- edge message kernel: filter WMMA + gather + atomic scatter ----------------
// One wave handles 16 edges. A = phi*fcut (16x32, K=20 zero-padded), B = packed filt_W_t.
__global__ __launch_bounds__(32) void edge_kernel(const float* __restrict__ r_ij,
                                                  const int* __restrict__ idx_i,
                                                  const int* __restrict__ idx_j,
                                                  const unsigned short* __restrict__ Fp,
                                                  const float* __restrict__ fb,
                                                  const float* __restrict__ x,
                                                  const float* __restrict__ muOld,
                                                  float* __restrict__ q,
                                                  float* __restrict__ muNew) {
  __shared__ float sd[16], sfc[16], sdir[16][3];
  __shared__ int si[16], sj[16];
  const int lane = threadIdx.x;
  const int e0 = blockIdx.x * 16;
  if (lane < 16) {
    int e = e0 + lane;
    float rx = r_ij[(size_t)e * 3], ry = r_ij[(size_t)e * 3 + 1], rz = r_ij[(size_t)e * 3 + 2];
    float d = sqrtf(rx * rx + ry * ry + rz * rz);
    float inv = 1.0f / d;
    sd[lane] = d;
    sdir[lane][0] = rx * inv; sdir[lane][1] = ry * inv; sdir[lane][2] = rz * inv;
    sfc[lane] = (d < 5.0f) ? 0.5f * (__cosf(d * (3.14159265358979f / 5.0f)) + 1.0f) : 0.0f;
    si[lane] = idx_i[e];
    sj[lane] = idx_j[e];
  }
  __syncthreads();
  const int r = lane & 15, half = lane >> 4;
  const float d = sd[r], fc = sfc[r];
  const float W0 = 5.0f / 19.0f, INVW = 19.0f / 5.0f;
  u16x16 ua;
  const int base = half * 8;
#pragma unroll
  for (int i = 0; i < 16; ++i) {
    int K = base + (i < 8 ? i : i + 8);
    float v = 0.0f;
    if (K < N_RBF) { float tt = (d - (float)K * W0) * INVW; v = __expf(-0.5f * tt * tt) * fc; }
    ua[i] = f2bf(v);
  }
  bf16x16 afrag = __builtin_bit_cast(bf16x16, ua);

  // chunk 1: dq (filter cols 0..127)
#pragma unroll
  for (int ct = 0; ct < 8; ++ct) {
    int col = ct * 16 + r;
    f32x8 acc = {0.f, 0.f, 0.f, 0.f, 0.f, 0.f, 0.f, 0.f};
    acc = wmma_bf16(afrag, loadB(Fp, ct, lane), acc);
    float bias = fb[col];
#pragma unroll
    for (int v = 0; v < 8; ++v) {
      int el = v + half * 8;                       // local edge for this C element
      float filt = acc[v] + sfc[el] * bias;        // filters = phi_fc@W + fcut*b
      int j = sj[el], ii = si[el];
      float m = filt * x[(size_t)j * 384 + col];
      atomicAdd(&q[(size_t)ii * 128 + col], m);
    }
  }
  // chunks 2+3 paired (same feature f): dmu = dmuR*dir + dmumu*mu_old[j]
#pragma unroll
  for (int ct = 8; ct < 16; ++ct) {
    int colR = ct * 16 + r;        // 128..255
    int colM = colR + 128;         // 256..383
    int f = colR - 128;
    f32x8 z8 = {0.f, 0.f, 0.f, 0.f, 0.f, 0.f, 0.f, 0.f};
    f32x8 accR = wmma_bf16(afrag, loadB(Fp, ct,     lane), z8);
    f32x8 accM = wmma_bf16(afrag, loadB(Fp, ct + 8, lane), z8);
    float biasR = fb[colR], biasM = fb[colM];
#pragma unroll
    for (int v = 0; v < 8; ++v) {
      int el = v + half * 8;
      int j = sj[el], ii = si[el];
      float mR = (accR[v] + sfc[el] * biasR) * x[(size_t)j * 384 + colR];
      float mM = (accM[v] + sfc[el] * biasM) * x[(size_t)j * 384 + colM];
#pragma unroll
      for (int dd = 0; dd < 3; ++dd) {
        float contrib = mR * sdir[el][dd] + mM * muOld[((size_t)j * 3 + dd) * 128 + f];
        atomicAdd(&muNew[((size_t)ii * 3 + dd) * 128 + f], contrib);
      }
    }
  }
}

// ---------------- field interaction elementwise update ----------------
__global__ void field_update_kernel(const float* __restrict__ a_s, const float* __restrict__ a_v,
                                    const float* __restrict__ Eat, float* __restrict__ mu) {
  int tid = blockIdx.x * blockDim.x + threadIdx.x;   // N*128
  int n = tid >> 7, f = tid & 127;
  float as = a_s[tid];
  float e0 = Eat[n * 3], e1 = Eat[n * 3 + 1], e2 = Eat[n * 3 + 2];
  size_t b = (size_t)n * 3 * 128 + f;
  float v0 = a_v[b], v1 = a_v[b + 128], v2 = a_v[b + 256];
  float dot = v0 * e0 + v1 * e1 + v2 * e2;
  mu[b]       += as * e0 - dot * v0;
  mu[b + 128] += as * e1 - dot * v1;
  mu[b + 256] += as * e2 - dot * v2;
}

// ---------------- mixing: norms, ctx build, scal ----------------
__global__ void mixnorm_kernel(const float* __restrict__ mm, const float* __restrict__ q,
                               float* __restrict__ ctx, float* __restrict__ scal) {
  int tid = blockIdx.x * blockDim.x + threadIdx.x;   // N*128
  int n = tid >> 7, f = tid & 127;
  size_t b = (size_t)n * 3 * 256 + f;
  float V0 = mm[b], V1 = mm[b + 256], V2 = mm[b + 512];
  float W0 = mm[b + 128], W1 = mm[b + 384], W2 = mm[b + 640];
  float Vn = sqrtf(V0 * V0 + V1 * V1 + V2 * V2 + 1e-8f);
  ctx[(size_t)n * 256 + f]       = q[tid];
  ctx[(size_t)n * 256 + 128 + f] = Vn;
  scal[tid] = V0 * W0 + V1 * W1 + V2 * W2;
}

// ---------------- mixing final update ----------------
__global__ void mix_update_kernel(const float* __restrict__ y, const float* __restrict__ scal,
                                  const float* __restrict__ mm, float* __restrict__ q,
                                  float* __restrict__ mu) {
  int tid = blockIdx.x * blockDim.x + threadIdx.x;   // N*128
  int n = tid >> 7, f = tid & 127;
  size_t yb = (size_t)n * 384;
  float dq = y[yb + f], dmu = y[yb + 128 + f], dqmu = y[yb + 256 + f];
  q[tid] += dq + dqmu * scal[tid];
  size_t mb = (size_t)n * 3 * 128 + f;
  size_t wb = (size_t)n * 3 * 256 + 128 + f;
  mu[mb]       += dmu * mm[wb];
  mu[mb + 128] += dmu * mm[wb + 256];
  mu[mb + 256] += dmu * mm[wb + 512];
}

// ---------------- pack [q, mu] -> out [N,4,F] ----------------
__global__ void writeout_kernel(const float* __restrict__ q, const float* __restrict__ mu,
                                float* __restrict__ out) {
  int tid = blockIdx.x * blockDim.x + threadIdx.x;   // N*512
  int n = tid >> 9, c = (tid >> 7) & 3, f = tid & 127;
  out[tid] = (c == 0) ? q[(size_t)n * 128 + f] : mu[((size_t)n * 3 + (c - 1)) * 128 + f];
}

// ---------------- host launch ----------------
extern "C" void kernel_launch(void* const* d_in, const int* in_sizes, int n_in,
                              void* d_out, int out_size, void* d_ws, size_t ws_size,
                              hipStream_t stream) {
  (void)in_sizes; (void)n_in; (void)out_size; (void)ws_size;
  const int*   atomic_numbers = (const int*)  d_in[0];
  const float* r_ij           = (const float*)d_in[1];
  const int*   idx_i          = (const int*)  d_in[2];
  const int*   idx_j          = (const int*)  d_in[3];
  const int*   idx_m          = (const int*)  d_in[4];
  const float* e_field        = (const float*)d_in[5];
  const float* emb            = (const float*)d_in[6];
  const float* filt_W         = (const float*)d_in[7];
  const float* filt_b         = (const float*)d_in[8];
  const float* inter_W1       = (const float*)d_in[9];
  const float* inter_b1       = (const float*)d_in[10];
  const float* inter_W2       = (const float*)d_in[11];
  const float* inter_b2       = (const float*)d_in[12];
  const float* field_sW1      = (const float*)d_in[13];
  const float* field_sb1      = (const float*)d_in[14];
  const float* field_sW2      = (const float*)d_in[15];
  const float* field_sb2      = (const float*)d_in[16];
  const float* field_vW       = (const float*)d_in[17];
  const float* mix_muW        = (const float*)d_in[18];
  const float* mix_W1         = (const float*)d_in[19];
  const float* mix_b1         = (const float*)d_in[20];
  const float* mix_W2         = (const float*)d_in[21];
  const float* mix_b2         = (const float*)d_in[22];

  char* ws = (char*)d_ws;
  size_t off = 0;
  auto arena = [&](size_t bytes) -> char* {
    char* p = ws + off;
    off += (bytes + 255) & ~(size_t)255;
    return p;
  };
  float* q    = (float*)arena((size_t)N_ATOMS * 128 * 4);
  float* muA  = (float*)arena((size_t)N_ATOMS * 384 * 4);
  float* muB  = (float*)arena((size_t)N_ATOMS * 384 * 4);
  float* big  = (float*)arena((size_t)N_ATOMS * 768 * 4);  // x / a_v / mu_mix (time-shared)
  float* a_s  = (float*)arena((size_t)N_ATOMS * 128 * 4);
  float* ctx  = (float*)arena((size_t)N_ATOMS * 256 * 4);
  float* scal = (float*)arena((size_t)N_ATOMS * 128 * 4);
  float* yb   = (float*)arena((size_t)N_ATOMS * 384 * 4);
  float* Eat  = (float*)arena((size_t)N_ATOMS * 3 * 4);

  unsigned short *fWp[3], *iW1p[3], *iW2p[3], *sW1p[3], *sW2p[3], *vWp[3], *muWp[3], *mxW1p[3], *mxW2p[3];
  for (int t = 0; t < 3; ++t) {
    fWp[t]   = (unsigned short*)arena(1 * 24 * 512 * 2);
    iW1p[t]  = (unsigned short*)arena(4 *  8 * 512 * 2);
    iW2p[t]  = (unsigned short*)arena(4 * 24 * 512 * 2);
    sW1p[t]  = (unsigned short*)arena(4 *  8 * 512 * 2);
    sW2p[t]  = (unsigned short*)arena(4 *  8 * 512 * 2);
    vWp[t]   = (unsigned short*)arena(4 *  8 * 512 * 2);
    muWp[t]  = (unsigned short*)arena(4 * 16 * 512 * 2);
    mxW1p[t] = (unsigned short*)arena(8 *  8 * 512 * 2);
    mxW2p[t] = (unsigned short*)arena(4 * 24 * 512 * 2);
  }

  auto pack = [&](const float* src, int ld, int col0, int Kreal, int Ktiles, int Ntiles,
                  unsigned short* dst) {
    int total = Ktiles * Ntiles * 512;
    pack_kernel<<<(total + 255) / 256, 256, 0, stream>>>(src, ld, col0, Kreal, Ktiles, Ntiles, dst);
  };
  for (int t = 0; t < 3; ++t) {
    pack(filt_W,                    1152, t * 384,  20, 1, 24, fWp[t]);
    pack(inter_W1  + (size_t)t * 128 * 128, 128, 0, 128, 4,  8, iW1p[t]);
    pack(inter_W2  + (size_t)t * 128 * 384, 384, 0, 128, 4, 24, iW2p[t]);
    pack(field_sW1 + (size_t)t * 128 * 128, 128, 0, 128, 4,  8, sW1p[t]);
    pack(field_sW2 + (size_t)t * 128 * 128, 128, 0, 128, 4,  8, sW2p[t]);
    pack(field_vW  + (size_t)t * 128 * 128, 128, 0, 128, 4,  8, vWp[t]);
    pack(mix_muW   + (size_t)t * 128 * 256, 256, 0, 128, 4, 16, muWp[t]);
    pack(mix_W1    + (size_t)t * 256 * 128, 128, 0, 256, 8,  8, mxW1p[t]);
    pack(mix_W2    + (size_t)t * 128 * 384, 384, 0, 128, 4, 24, mxW2p[t]);
  }

  init_kernel<<<(N_ATOMS * 128) / 256, 256, 0, stream>>>(atomic_numbers, idx_m, emb, e_field,
                                                         q, muA, Eat);

  float* muCur = muA;
  float* muOther = muB;
  for (int t = 0; t < 3; ++t) {
    // x = silu(q@W1+b1)@W2+b2  [N,384]
    mlp16_kernel<128><<<N_ATOMS / 16, 32, 0, stream>>>(q, iW1p[t], inter_b1 + t * 128,
                                                       iW2p[t], inter_b2 + t * 384, 384, big);
    // snapshot mu (messages must read pre-update mu)
    hipMemcpyAsync(muOther, muCur, (size_t)N_ATOMS * 384 * 4, hipMemcpyDeviceToDevice, stream);
    edge_kernel<<<N_EDGES / 16, 32, 0, stream>>>(r_ij, idx_i, idx_j, fWp[t], filt_b + t * 384,
                                                 big, muCur, q, muOther);
    { float* tmp = muCur; muCur = muOther; muOther = tmp; }
    // field interaction
    mlp16_kernel<128><<<N_ATOMS / 16, 32, 0, stream>>>(q, sW1p[t], field_sb1 + t * 128,
                                                       sW2p[t], field_sb2 + t * 128, 128, a_s);
    linear16_kernel<<<(N_ATOMS * 3) / 16, 32, 0, stream>>>(muCur, vWp[t], 128, big);  // a_v
    field_update_kernel<<<(N_ATOMS * 128) / 256, 256, 0, stream>>>(a_s, big, Eat, muCur);
    // mixing
    linear16_kernel<<<(N_ATOMS * 3) / 16, 32, 0, stream>>>(muCur, muWp[t], 256, big); // mu_mix
    mixnorm_kernel<<<(N_ATOMS * 128) / 256, 256, 0, stream>>>(big, q, ctx, scal);
    mlp16_kernel<256><<<N_ATOMS / 16, 32, 0, stream>>>(ctx, mxW1p[t], mix_b1 + t * 128,
                                                       mxW2p[t], mix_b2 + t * 384, 384, yb);
    mix_update_kernel<<<(N_ATOMS * 128) / 256, 256, 0, stream>>>(yb, scal, big, q, muCur);
  }
  writeout_kernel<<<(N_ATOMS * 512) / 256, 256, 0, stream>>>(q, muCur, (float*)d_out);
}
